// VQVAE_4002909520242
// MI455X (gfx1250) — matrix operand order repaired and checked by reading, not compiled, not run on hardware
//
#include <hip/hip_runtime.h>
#include <hip/hip_bf16.h>

typedef __attribute__((ext_vector_type(16))) _Float16 v16h;
typedef __attribute__((ext_vector_type(8)))  float    v8f;
typedef __attribute__((ext_vector_type(4)))  _Float16 v4h;
typedef __attribute__((ext_vector_type(2)))  _Float16 v2h;
typedef __attribute__((ext_vector_type(4)))  unsigned int v4u;
typedef __attribute__((ext_vector_type(4)))  float    v4f;
typedef __attribute__((ext_vector_type(2)))  float    v2f;

// GCC-style int4 vector in explicit address spaces, matching the builtin's
// parameter types: (v4i AS1* src, v4i AS3* dst, imm offset, imm cpol)
typedef int v4ig __attribute__((vector_size(16)));
typedef __attribute__((address_space(1))) v4ig as1_v4i;   // global
typedef __attribute__((address_space(3))) v4ig as3_v4i;   // LDS

#if defined(__HIP_DEVICE_COMPILE__) && \
    __has_builtin(__builtin_amdgcn_global_load_async_to_lds_b128)
#define ASYNC_LDS 1
#else
#define ASYNC_LDS 0
#endif

__device__ __forceinline__ void wait_async0() {
#if ASYNC_LDS
#if __has_builtin(__builtin_amdgcn_s_wait_asynccnt)
  __builtin_amdgcn_s_wait_asynccnt(0);
#else
  asm volatile("s_wait_asynccnt 0" ::: "memory");
#endif
#endif
}

union FragAB {            // one 16x32 (A) or 32x16 (B) f16 fragment per wave
  v16h h;
  v4u  q[2];
};

__device__ __forceinline__ v8f wmma_f16(v16h a, v16h b, v8f c) {
  // D = A(16x32,f16) * B(32x16,f16) + C(16x16,f32)
  return __builtin_amdgcn_wmma_f32_16x16x32_f16(
      /*neg_a=*/false, a, /*neg_b=*/false, b,
      /*c_mod=*/(short)0, c, /*reuse_a=*/false, /*reuse_b=*/false);
}

__device__ __forceinline__ v8f vzero8() {
  v8f z = {0.f, 0.f, 0.f, 0.f, 0.f, 0.f, 0.f, 0.f};
  return z;
}

enum : int { EPI_RELU_F16 = 0, EPI_ZE = 1, EPI_SIG_F32 = 2 };

// C = epilogue(A[M,K] * Bw[K,N] + bias[N])
// A is f32 or f16 (converted to f16 into LDS); Bw/bias are f32.
// Software-pipelined: tile k+1 global->VGPR loads overlap tile k WMMAs.
template<typename TA, int EPI, int BM, int BN,
         int WAVE_M, int WAVE_N, int WM_FR, int WN_FR>
__global__ __launch_bounds__(WAVE_M * WAVE_N * 32)
void gemm_wmma_kernel(const TA* __restrict__ A, const float* __restrict__ Bw,
                      const float* __restrict__ bias,
                      void* __restrict__ out0, _Float16* __restrict__ out1,
                      int M, int N, int K, int lda, int ldb, int ldo)
{
  constexpr int BK  = 64;       // two WMMA k-steps per LDS tile
  constexpr int NT  = WAVE_M * WAVE_N * 32;
  constexpr int LDA = BK + 8;   // 144B row stride: conflict-free b128 loads
  constexpr int LDB = BK + 8;
  constexpr int AIT = (BM * BK / 4) / NT;
  constexpr int BIT = (BN * BK / 4) / NT;
  static_assert(BM == WAVE_M * WM_FR * 16 && BN == WAVE_N * WN_FR * 16, "tile");
  static_assert((BM * BK / 4) % NT == 0 && (BN * BK / 4) % NT == 0, "uniform");

  __shared__ alignas(16) _Float16 As[BM * LDA];
  __shared__ alignas(16) _Float16 Bs[BN * LDB];

  const int tid  = threadIdx.x;
  const int lane = tid & 31;
  const int wave = tid >> 5;
  const int wm   = wave % WAVE_M;
  const int wn   = wave / WAVE_M;
  const int l15  = lane & 15;
  const int hsel = lane >> 4;
  const int mb   = blockIdx.y * BM;
  const int nb   = blockIdx.x * BN;

  v8f acc[WM_FR][WN_FR];
#pragma unroll
  for (int i = 0; i < WM_FR; ++i)
#pragma unroll
    for (int j = 0; j < WN_FR; ++j) acc[i][j] = vzero8();

  v4h aReg[AIT];
  v4h bReg[BIT];

  auto loadTiles = [&](int k0) {
    const bool fullK = (k0 + BK) <= K;
#pragma unroll
    for (int it = 0; it < AIT; ++it) {
      int i  = tid + it * NT;
      int r  = i / (BK / 4);
      int kk = (i % (BK / 4)) * 4;
      v4h hv;
      if (fullK) {
        const TA* src = A + (size_t)(mb + r) * lda + (k0 + kk);
#pragma unroll
        for (int j = 0; j < 4; ++j) hv[j] = (_Float16)(float)src[j];
      } else {
#pragma unroll
        for (int j = 0; j < 4; ++j) {
          int gk = k0 + kk + j;
          hv[j] = (gk < K) ? (_Float16)(float)A[(size_t)(mb + r) * lda + gk]
                           : (_Float16)0.f;
        }
      }
      aReg[it] = hv;
    }
    const bool fullB = fullK && ((nb + BN) <= N);
#pragma unroll
    for (int it = 0; it < BIT; ++it) {
      int i  = tid + it * NT;
      int n  = i % BN;                 // consecutive lanes -> consecutive n
      int kk = (i / BN) * 4;
      v4h hv;
      if (fullB) {
#pragma unroll
        for (int j = 0; j < 4; ++j)
          hv[j] = (_Float16)Bw[(size_t)(k0 + kk + j) * ldb + (nb + n)];
      } else {
#pragma unroll
        for (int j = 0; j < 4; ++j) {
          int gk = k0 + kk + j, gn = nb + n;
          hv[j] = (gk < K && gn < N) ? (_Float16)Bw[(size_t)gk * ldb + gn]
                                     : (_Float16)0.f;
        }
      }
      bReg[it] = hv;
    }
  };

  auto storeTiles = [&]() {
#pragma unroll
    for (int it = 0; it < AIT; ++it) {
      int i  = tid + it * NT;
      int r  = i / (BK / 4);
      int kk = (i % (BK / 4)) * 4;
      *(v4h*)&As[r * LDA + kk] = aReg[it];
    }
#pragma unroll
    for (int it = 0; it < BIT; ++it) {
      int i  = tid + it * NT;
      int n  = i % BN;
      int kk = (i / BN) * 4;
      *(v4h*)&Bs[n * LDB + kk] = bReg[it];
    }
  };

  auto compute = [&]() {
#pragma unroll
    for (int s = 0; s < 2; ++s) {           // two 16x16x32 k-steps per tile
      FragAB a[WM_FR], b[WN_FR];
      const int aklo = s * 32 + hsel * 8;
#pragma unroll
      for (int fm = 0; fm < WM_FR; ++fm) {
        int row = wm * WM_FR * 16 + fm * 16 + l15;
        const _Float16* p = &As[row * LDA + aklo];
        a[fm].q[0] = *(const v4u*)(p);
        a[fm].q[1] = *(const v4u*)(p + 16);
      }
      const int bklo = s * 32 + hsel * 8;
#pragma unroll
      for (int fn = 0; fn < WN_FR; ++fn) {
        int col = wn * WN_FR * 16 + fn * 16 + l15;
        const _Float16* p = &Bs[col * LDB + bklo];
        b[fn].q[0] = *(const v4u*)(p);
        b[fn].q[1] = *(const v4u*)(p + 16);
      }
#pragma unroll
      for (int fm = 0; fm < WM_FR; ++fm)
#pragma unroll
        for (int fn = 0; fn < WN_FR; ++fn)
          acc[fm][fn] = wmma_f16(a[fm].h, b[fn].h, acc[fm][fn]);
    }
  };

  // prologue
  loadTiles(0);
  storeTiles();
  __syncthreads();
  for (int k0 = 0; k0 < K; k0 += BK) {
    const int kn = k0 + BK;
    if (kn < K) loadTiles(kn);      // overlap next tile loads with WMMAs
    compute();
    if (kn < K) {
      __syncthreads();
      storeTiles();
      __syncthreads();
    }
  }

  // ---- epilogue: C layout lane=(N), vgpr r=(M row 0..7 / 8..15) ----
#pragma unroll
  for (int fm = 0; fm < WM_FR; ++fm) {
#pragma unroll
    for (int fn = 0; fn < WN_FR; ++fn) {
      int gn = nb + wn * WN_FR * 16 + fn * 16 + l15;
      if (gn >= N) continue;
      int gm0 = mb + wm * WM_FR * 16 + fm * 16 + hsel * 8;
      float bs = bias[gn];
#pragma unroll
      for (int r = 0; r < 8; ++r) {
        float v  = acc[fm][fn][r] + bs;
        size_t o = (size_t)(gm0 + r) * ldo + gn;
        if constexpr (EPI == EPI_RELU_F16) {
          ((_Float16*)out0)[o] = (_Float16)fmaxf(v, 0.f);
        } else if constexpr (EPI == EPI_ZE) {
          ((float*)out0)[o] = v;                     // z_e (f32 output)
          out1[o] = (_Float16)(-2.f * v);            // -2*z_e for distance GEMM
        } else {
          ((float*)out0)[o] = 1.f / (1.f + __expf(-v));  // sigmoid
        }
      }
    }
  }
}

// codebook -> f16 copy + per-entry squared norms
__global__ void emb_prep_kernel(const float* __restrict__ emb,
                                _Float16* __restrict__ emb16,
                                float* __restrict__ nrm, int KE, int D)
{
  int e = blockIdx.x * blockDim.x + threadIdx.x;
  if (e >= KE) return;
  const float* src = emb + (size_t)e * D;
  _Float16* dst = emb16 + (size_t)e * D;
  float s = 0.f;
  for (int d = 0; d < D; d += 4) {
    v4f v = *(const v4f*)(src + d);
    s += v.x * v.x + v.y * v.y + v.z * v.z + v.w * v.w;
    v4h h = {(_Float16)v.x, (_Float16)v.y, (_Float16)v.z, (_Float16)v.w};
    *(v4h*)(dst + d) = h;
  }
  nrm[e] = s;
}

// argmin_k( -2*z_e . e_k + ||e_k||^2 ) via WMMA, then gather z_q.
__global__ __launch_bounds__(128)
void vq_argmin_kernel(const _Float16* __restrict__ zem2,   // [B][64] = -2*z_e
                      const _Float16* __restrict__ emb16,  // [KE][64]
                      const float* __restrict__ embnrm,    // [KE]
                      const float* __restrict__ emb,       // [KE][64] f32
                      float* __restrict__ zq_out,          // [B][64] f32
                      _Float16* __restrict__ zq16,         // [B][64] f16
                      int KE)
{
  constexpr int CH  = 128;   // codebook entries staged per chunk
  constexpr int ELD = 72;    // padded f16 row stride (144B, conflict-free)
  __shared__ alignas(16) _Float16 Es[CH * ELD];
  __shared__ float Ns[CH];

  const int tid  = threadIdx.x;
  const int lane = tid & 31;
  const int wave = tid >> 5;
  const int l15  = lane & 15;
  const int hsel = lane >> 4;
  const int m0   = (blockIdx.x * 4 + wave) * 16;   // 16 rows per wave

  // A fragments: -2*z_e rows m0..m0+15, latent K=0..63 (two 16x32 frags)
  FragAB afr[2];
  {
    int row = m0 + l15;
#pragma unroll
    for (int s = 0; s < 2; ++s) {
      int kb = s * 32 + hsel * 8;
      const _Float16* p = zem2 + (size_t)row * 64 + kb;
      afr[s].q[0] = *(const v4u*)(p);
      afr[s].q[1] = *(const v4u*)(p + 16);
    }
  }

  float minv[8];
  int   mini[8];
#pragma unroll
  for (int r = 0; r < 8; ++r) { minv[r] = 3.4e38f; mini[r] = 0; }

  for (int c0 = 0; c0 < KE; c0 += CH) {
    // stage 128 codebook entries (f16) + norms into LDS
#pragma unroll 1
    for (int i = tid; i < CH * 8; i += 128) {
      int e = i >> 3, q = i & 7;
#if ASYNC_LDS
      // ASYNCcnt-tracked direct global->LDS copy (no VGPR round trip)
      __builtin_amdgcn_global_load_async_to_lds_b128(
          (as1_v4i*)(emb16 + (size_t)(c0 + e) * 64 + q * 8),
          (as3_v4i*)&Es[e * ELD + q * 8],
          0, 0);
#else
      *(v4u*)&Es[e * ELD + q * 8] =
          *(const v4u*)(emb16 + (size_t)(c0 + e) * 64 + q * 8);
#endif
    }
    if (tid < CH) Ns[tid] = embnrm[c0 + tid];
    wait_async0();
    __syncthreads();

#pragma unroll 1
    for (int t = 0; t < CH / 16; ++t) {
      v8f accv = vzero8();
      int eloc = t * 16 + l15;
#pragma unroll
      for (int s = 0; s < 2; ++s) {
        FragAB bfr;
        const _Float16* p = &Es[eloc * ELD + s * 32 + hsel * 8];
        bfr.q[0] = *(const v4u*)(p);
        bfr.q[1] = *(const v4u*)(p + 16);
        accv = wmma_f16(afr[s].h, bfr.h, accv);
      }
      float nrm = Ns[t * 16 + l15];
      int gidx  = c0 + t * 16 + l15;
#pragma unroll
      for (int r = 0; r < 8; ++r) {
        float sc = accv[r] + nrm;
        if (sc < minv[r]) { minv[r] = sc; mini[r] = gidx; }
      }
    }
    __syncthreads();
  }

  // reduce across the 16 lanes holding different column stripes of each row
#pragma unroll
  for (int r = 0; r < 8; ++r) {
#pragma unroll
    for (int m = 8; m >= 1; m >>= 1) {
      float ov = __shfl_xor(minv[r], m, 32);
      int   oi = __shfl_xor(mini[r], m, 32);
      if (ov < minv[r] || (ov == minv[r] && oi < mini[r])) {
        minv[r] = ov; mini[r] = oi;
      }
    }
  }

  // gather z_q rows cooperatively (f32 out + f16 for decoder GEMM)
#pragma unroll
  for (int r = 0; r < 16; ++r) {
    int k  = __shfl(mini[r & 7], (r < 8) ? 0 : 16, 32);
    int gm = m0 + r;
    v2f v = *(const v2f*)(emb + (size_t)k * 64 + lane * 2);
    *(v2f*)(zq_out + (size_t)gm * 64 + lane * 2) = v;
    v2h h = {(_Float16)v.x, (_Float16)v.y};
    *(v2h*)(zq16 + (size_t)gm * 64 + lane * 2) = h;
  }
}

extern "C" void kernel_launch(void* const* d_in, const int* in_sizes, int n_in,
                              void* d_out, int out_size, void* d_ws, size_t ws_size,
                              hipStream_t stream)
{
  const int Bn = 32768, DIN = 784, DH = 1024, DLAT = 64, KE = 8192;

  const float* x   = (const float*)d_in[0];
  const float* W1  = (const float*)d_in[1];
  const float* b1  = (const float*)d_in[2];
  const float* W2  = (const float*)d_in[3];
  const float* b2  = (const float*)d_in[4];
  const float* W3  = (const float*)d_in[5];
  const float* b3  = (const float*)d_in[6];
  const float* W4  = (const float*)d_in[7];
  const float* b4  = (const float*)d_in[8];
  const float* emb = (const float*)d_in[9];

  float* xrec = (float*)d_out;                    // [B,784]
  float* ze   = xrec + (size_t)Bn * DIN;          // [B,64]
  float* zq   = ze   + (size_t)Bn * DLAT;         // [B,64]

  // workspace layout (h1 and h3 share the first buffer)
  char* ws = (char*)d_ws;
  const size_t OFF_H    = 0;                                   // B*DH f16
  const size_t OFF_ZEM2 = OFF_H    + (size_t)Bn * DH * 2;      // B*DLAT f16
  const size_t OFF_ZQ16 = OFF_ZEM2 + (size_t)Bn * DLAT * 2;    // B*DLAT f16
  const size_t OFF_E16  = OFF_ZQ16 + (size_t)Bn * DLAT * 2;    // KE*DLAT f16
  const size_t OFF_NRM  = OFF_E16  + (size_t)KE * DLAT * 2;    // KE f32
  _Float16* h16   = (_Float16*)(ws + OFF_H);
  _Float16* zem2  = (_Float16*)(ws + OFF_ZEM2);
  _Float16* zq16  = (_Float16*)(ws + OFF_ZQ16);
  _Float16* emb16 = (_Float16*)(ws + OFF_E16);
  float*    enrm  = (float*)   (ws + OFF_NRM);

  // 0) codebook f16 + norms
  emb_prep_kernel<<<KE / 256, 256, 0, stream>>>(emb, emb16, enrm, KE, DLAT);

  // 1) h1 = relu(x @ W1 + b1)  -> f16   [M=32768,N=1024,K=784]
  gemm_wmma_kernel<float, EPI_RELU_F16, 128, 128, 2, 4, 4, 2>
      <<<dim3(DH / 128, Bn / 128), 256, 0, stream>>>(
          x, W1, b1, (void*)h16, nullptr, Bn, DH, DIN, DIN, DH, DH);

  // 2) z_e = h1 @ W2 + b2 (f32 out) ; zem2 = -2*z_e (f16)  [N=64,K=1024]
  gemm_wmma_kernel<_Float16, EPI_ZE, 128, 64, 2, 2, 4, 2>
      <<<dim3(1, Bn / 128), 128, 0, stream>>>(
          h16, W2, b2, (void*)ze, zem2, Bn, DLAT, DH, DH, DLAT, DLAT);

  // 3) nearest codebook entry + gather z_q (f32 out + f16 copy)
  vq_argmin_kernel<<<Bn / 64, 128, 0, stream>>>(zem2, emb16, enrm, emb,
                                                zq, zq16, KE);

  // 4) h3 = relu(z_q @ W3 + b3) -> f16 (reuses h16)  [N=1024,K=64]
  gemm_wmma_kernel<_Float16, EPI_RELU_F16, 128, 128, 2, 4, 4, 2>
      <<<dim3(DH / 128, Bn / 128), 256, 0, stream>>>(
          zq16, W3, b3, (void*)h16, nullptr, Bn, DH, DLAT, DLAT, DH, DH);

  // 5) x_recon = sigmoid(h3 @ W4 + b4) -> f32  [N=784,K=1024]
  gemm_wmma_kernel<_Float16, EPI_SIG_F32, 128, 128, 2, 4, 4, 2>
      <<<dim3((DIN + 127) / 128, Bn / 128), 256, 0, stream>>>(
          h16, W4, b4, (void*)xrec, nullptr, Bn, DIN, DH, DH, DIN, DIN);
}
